// SimpleAttention_41824391528678
// MI455X (gfx1250) — matrix operand order repaired and checked by reading, not compiled, not run on hardware
//
#include <hip/hip_runtime.h>
#include <hip/hip_bf16.h>

typedef __attribute__((ext_vector_type(16))) __bf16 v16bf;
typedef __attribute__((ext_vector_type(8)))  __bf16 v8bf;
typedef __attribute__((ext_vector_type(8)))  float  v8f;

#define BATCH  4
#define S_LEN  8192
#define FDIM   1024
#define HEADS  16
#define DDIM   64
#define NCHUNK 8
#define BK     64

__device__ __forceinline__ v16bf cat8(v8bf lo, v8bf hi) {
  v16bf r;
#pragma unroll
  for (int i = 0; i < 8; ++i) { r[i] = lo[i]; r[i + 8] = hi[i]; }
  return r;
}

__device__ __forceinline__ v8f wmma_bf16(v16bf a, v16bf b, v8f c) {
  return __builtin_amdgcn_wmma_f32_16x16x32_bf16(false, a, false, b, (short)0, c,
                                                 false, false);
}

// ---------------------------------------------------------------------------
// Kernel A: Y(32768x1024) = X(32768x1024) @ W(1024x1024), fp32 in/out,
// bf16 WMMA inner. Block tile 128x128, BK=64, double-buffered LDS, one
// barrier per K-iteration, packed b128 LDS staging (incl. transposed W tile).
// ---------------------------------------------------------------------------
__global__ __launch_bounds__(256)
void proj_gemm_kernel(const float* __restrict__ X, const float* __restrict__ W,
                      float* __restrict__ Y)
{
  // pitch 72 bf16 (144 B = 36 dwords): conflict-free 16-row fragment reads
  __shared__ __attribute__((aligned(32))) __bf16 As[2][128][72];  // [m][k]
  __shared__ __attribute__((aligned(32))) __bf16 Bs[2][128][72];  // [n][k] (W^T)

  const int tid  = threadIdx.x;
  const int lane = tid & 31, wave = tid >> 5;
  const int wm = wave >> 1, wn = wave & 1;          // 4 x 2 wave grid
  const int m0 = blockIdx.y * 128;
  const int n0 = blockIdx.x * 128;
  const int l15 = lane & 15, kh = lane >> 4;

  // A staging: thread covers 1 row x 32 k           (128 rows x 2 threads)
  const int arow = tid >> 1;
  const int acol = (tid & 1) * 32;
  // B staging: thread covers 8 k x 4 n sub-block    (8 x 32 threads)
  const int bkq = (tid >> 5) * 8;      // 0..56 (k)
  const int bnq = (tid & 31) * 4;      // 0..124 (n)

  float4 aReg[8], bReg[8];

  auto loadTiles = [&](int k0) {
    const float* xs = X + (size_t)(m0 + arow) * FDIM + k0 + acol;
#pragma unroll
    for (int j = 0; j < 8; ++j) aReg[j] = *(const float4*)(xs + j * 4);
#pragma unroll
    for (int i = 0; i < 8; ++i)
      bReg[i] = *(const float4*)(W + (size_t)(k0 + bkq + i) * FDIM + n0 + bnq);
  };

  auto storeTiles = [&](int buf) {
    // A: pack pairs of float4 -> v8bf, 4 x b128 stores along k
#pragma unroll
    for (int j = 0; j < 8; j += 2) {
      v8bf p;
      p[0] = (__bf16)aReg[j].x;     p[1] = (__bf16)aReg[j].y;
      p[2] = (__bf16)aReg[j].z;     p[3] = (__bf16)aReg[j].w;
      p[4] = (__bf16)aReg[j + 1].x; p[5] = (__bf16)aReg[j + 1].y;
      p[6] = (__bf16)aReg[j + 1].z; p[7] = (__bf16)aReg[j + 1].w;
      *(v8bf*)&As[buf][arow][acol + j * 4] = p;
    }
    // B: transpose in registers; for each of 4 n, pack 8 bf16 along k -> b128
#pragma unroll
    for (int j = 0; j < 4; ++j) {
      v8bf p;
#pragma unroll
      for (int i = 0; i < 8; ++i)
        p[i] = (__bf16)(((const float*)&bReg[i])[j]);
      *(v8bf*)&Bs[buf][bnq + j][bkq] = p;
    }
  };

  v8f acc[2][4] = {};

  loadTiles(0);
  storeTiles(0);
  __syncthreads();

  int buf = 0;
  for (int k0 = 0; k0 < FDIM; k0 += BK) {
    const bool more = (k0 + BK) < FDIM;
    if (more) {
      loadTiles(k0 + BK);  // global loads in flight across the WMMA block
      __builtin_prefetch(X + (size_t)(m0 + arow) * FDIM + k0 + 2 * BK + acol, 0, 1);
    }

#pragma unroll
    for (int ks = 0; ks < 2; ++ks) {
      v16bf afrag[2];
#pragma unroll
      for (int i = 0; i < 2; ++i) {
        const __bf16* row = &As[buf][wm * 32 + i * 16 + l15][ks * 32];
        afrag[i] = cat8(*(const v8bf*)(row + kh * 8),
                        *(const v8bf*)(row + 16 + kh * 8));
      }
#pragma unroll
      for (int j = 0; j < 4; ++j) {
        const __bf16* row = &Bs[buf][wn * 64 + j * 16 + l15][ks * 32];
        v16bf bfrag = cat8(*(const v8bf*)(row + kh * 8),
                           *(const v8bf*)(row + 16 + kh * 8));
#pragma unroll
        for (int i = 0; i < 2; ++i)
          acc[i][j] = wmma_bf16(afrag[i], bfrag, acc[i][j]);
      }
    }

    if (more) storeTiles(buf ^ 1);
    __syncthreads();
    buf ^= 1;
  }

#pragma unroll
  for (int i = 0; i < 2; ++i)
#pragma unroll
    for (int j = 0; j < 4; ++j) {
      const int col  = n0 + wn * 64 + j * 16 + l15;
      const int rowb = m0 + wm * 32 + i * 16 + kh * 8;
#pragma unroll
      for (int r = 0; r < 8; ++r)
        Y[(size_t)(rowb + r) * FDIM + col] = acc[i][j][r];
    }
}

// ---------------------------------------------------------------------------
// Kernel B: partial p_attn = K^T V over one 1024-row n-chunk, with per-row
// (64-wide) LayerNorm of K and V fused into the load. 16 waves, each owns one
// 16x16 tile of the 64x64 output; n is the WMMA K dim (step 32).
// ---------------------------------------------------------------------------
__global__ __launch_bounds__(512)
void ktv_partial_kernel(const float* __restrict__ Yk, const float* __restrict__ Yv,
                        const float* __restrict__ ksc, const float* __restrict__ kbi,
                        const float* __restrict__ vsc, const float* __restrict__ vbi,
                        float* __restrict__ partials)
{
  __shared__ __attribute__((aligned(32))) __bf16 KT[64][40];  // [d][n_local]
  __shared__ __attribute__((aligned(32))) __bf16 VT[64][40];  // [e][n_local]

  const int bh = blockIdx.x;
  const int b = bh >> 4, h = bh & 15;
  const int chunk = blockIdx.y;
  const float* Kb = Yk + (size_t)b * S_LEN * FDIM + (size_t)h * 512 * FDIM; // [8192][64]
  const float* Vb = Yv + (size_t)b * S_LEN * FDIM + (size_t)h * 512 * FDIM;

  const int tid  = threadIdx.x;
  const int lane = tid & 31, wave = tid >> 5;
  const int l15 = lane & 15, kh = lane >> 4;
  const int dt = wave >> 2, et = wave & 3;

  const int ncol = 2 * wave + kh;   // local n row this lane stages (0..31)
  const int dq   = l15 * 4;         // feature quad within the 64-wide LN group

  const float4 ks4 = *(const float4*)(ksc + h * DDIM + dq);
  const float4 kb4 = *(const float4*)(kbi + h * DDIM + dq);
  const float4 vs4 = *(const float4*)(vsc + h * DDIM + dq);
  const float4 vb4 = *(const float4*)(vbi + h * DDIM + dq);

  v8f acc = {};

  for (int iter = 0; iter < 32; ++iter) {
    const int n = chunk * 1024 + iter * 32 + ncol;
    {  // K row: LN over 64 contiguous floats via half-wave shfl_xor reduction
      float4 x = *(const float4*)(Kb + (size_t)n * DDIM + dq);
      float s = x.x + x.y + x.z + x.w;
      float q = x.x * x.x + x.y * x.y + x.z * x.z + x.w * x.w;
#pragma unroll
      for (int m = 1; m < 16; m <<= 1) {
        s += __shfl_xor(s, m, 32);
        q += __shfl_xor(q, m, 32);
      }
      float mu  = s * (1.0f / 64.0f);
      float var = q * (1.0f / 64.0f) - mu * mu;
      float ri  = rsqrtf(var + 1e-6f);
      KT[dq + 0][ncol] = (__bf16)((x.x - mu) * ri * ks4.x + kb4.x);
      KT[dq + 1][ncol] = (__bf16)((x.y - mu) * ri * ks4.y + kb4.y);
      KT[dq + 2][ncol] = (__bf16)((x.z - mu) * ri * ks4.z + kb4.z);
      KT[dq + 3][ncol] = (__bf16)((x.w - mu) * ri * ks4.w + kb4.w);
    }
    {  // V row
      float4 x = *(const float4*)(Vb + (size_t)n * DDIM + dq);
      float s = x.x + x.y + x.z + x.w;
      float q = x.x * x.x + x.y * x.y + x.z * x.z + x.w * x.w;
#pragma unroll
      for (int m = 1; m < 16; m <<= 1) {
        s += __shfl_xor(s, m, 32);
        q += __shfl_xor(q, m, 32);
      }
      float mu  = s * (1.0f / 64.0f);
      float var = q * (1.0f / 64.0f) - mu * mu;
      float ri  = rsqrtf(var + 1e-6f);
      VT[dq + 0][ncol] = (__bf16)((x.x - mu) * ri * vs4.x + vb4.x);
      VT[dq + 1][ncol] = (__bf16)((x.y - mu) * ri * vs4.y + vb4.y);
      VT[dq + 2][ncol] = (__bf16)((x.z - mu) * ri * vs4.z + vb4.z);
      VT[dq + 3][ncol] = (__bf16)((x.w - mu) * ri * vs4.w + vb4.w);
    }
    __syncthreads();

    const __bf16* arow = &KT[dt * 16 + l15][0];
    v16bf a = cat8(*(const v8bf*)(arow + kh * 8),
                   *(const v8bf*)(arow + 16 + kh * 8));
    const __bf16* brow = &VT[et * 16 + l15][0];
    v16bf bb = cat8(*(const v8bf*)(brow + kh * 8),
                    *(const v8bf*)(brow + 16 + kh * 8));
    acc = wmma_bf16(a, bb, acc);
    __syncthreads();
  }

  float* out = partials + ((size_t)bh * NCHUNK + chunk) * 4096;
#pragma unroll
  for (int r = 0; r < 8; ++r) {
    const int d = dt * 16 + kh * 8 + r;
    const int e = et * 16 + l15;
    out[d * 64 + e] = acc[r];
  }
}

// ---------------------------------------------------------------------------
// Kernel C: deterministic split-K reduction + 1/seq_len scale -> p_attn output
// ---------------------------------------------------------------------------
__global__ __launch_bounds__(256)
void reduce_pattn_kernel(const float* __restrict__ partials, float* __restrict__ p_out)
{
  const int i  = blockIdx.x * 256 + threadIdx.x;  // 262144 outputs
  const int bh = i >> 12;
  const int idx = i & 4095;
  float s = 0.f;
#pragma unroll
  for (int c = 0; c < NCHUNK; ++c)
    s += partials[((size_t)bh * NCHUNK + c) * 4096 + idx];
  p_out[i] = s * (1.0f / 8192.0f);
}

// ---------------------------------------------------------------------------
// Kernel D: x = Q @ p_attn per (b,h); in-place on the att_output region.
// Each wave stages its own 16 rows of Q into LDS, computes a 16x64 strip
// (4 n-tiles x 2 k-steps of bf16 WMMA), then overwrites those rows.
// ---------------------------------------------------------------------------
__global__ __launch_bounds__(256)
void q_pattn_kernel(float* __restrict__ att, const float* __restrict__ p_out)
{
  __shared__ __attribute__((aligned(32))) __bf16 PT[64][72];   // [e][d]
  __shared__ __attribute__((aligned(32))) __bf16 Qs[128][72];  // [row][d]

  const int bh = blockIdx.y;
  const int b = bh >> 4, h = bh & 15;
  float* Qb = att + (size_t)b * S_LEN * FDIM + (size_t)h * 512 * FDIM;  // [8192][64]
  const int n0 = blockIdx.x * 128;

  const int tid  = threadIdx.x;
  const int lane = tid & 31, wave = tid >> 5;
  const int l15 = lane & 15, kh = lane >> 4;

  {  // load p_attn (64x64) transposed into LDS as bf16
    const float* p = p_out + (size_t)bh * 4096;
    const int d  = tid >> 2;
    const int eb = (tid & 3) * 16;
#pragma unroll
    for (int j = 0; j < 16; j += 4) {
      float4 f = *(const float4*)(p + d * 64 + eb + j);
      PT[eb + j + 0][d] = (__bf16)f.x;
      PT[eb + j + 1][d] = (__bf16)f.y;
      PT[eb + j + 2][d] = (__bf16)f.z;
      PT[eb + j + 3][d] = (__bf16)f.w;
    }
  }
  {  // each wave loads its own 16 Q rows (packed b128 LDS stores)
    const int qrow = wave * 16 + (lane >> 1);
    const int qcb  = (lane & 1) * 32;
    const float* src = Qb + (size_t)(n0 + qrow) * DDIM + qcb;
#pragma unroll
    for (int j = 0; j < 32; j += 8) {
      float4 f0 = *(const float4*)(src + j);
      float4 f1 = *(const float4*)(src + j + 4);
      v8bf p;
      p[0] = (__bf16)f0.x; p[1] = (__bf16)f0.y;
      p[2] = (__bf16)f0.z; p[3] = (__bf16)f0.w;
      p[4] = (__bf16)f1.x; p[5] = (__bf16)f1.y;
      p[6] = (__bf16)f1.z; p[7] = (__bf16)f1.w;
      *(v8bf*)&Qs[qrow][qcb + j] = p;
    }
  }
  __syncthreads();

  v16bf a[2];
#pragma unroll
  for (int ks = 0; ks < 2; ++ks) {
    const __bf16* row = &Qs[wave * 16 + l15][ks * 32];
    a[ks] = cat8(*(const v8bf*)(row + kh * 8),
                 *(const v8bf*)(row + 16 + kh * 8));
  }
#pragma unroll
  for (int j = 0; j < 4; ++j) {
    v8f acc = {};
#pragma unroll
    for (int ks = 0; ks < 2; ++ks) {
      const __bf16* row = &PT[j * 16 + l15][ks * 32];
      v16bf bb = cat8(*(const v8bf*)(row + kh * 8),
                      *(const v8bf*)(row + 16 + kh * 8));
      acc = wmma_bf16(a[ks], bb, acc);
    }
    const int e = j * 16 + l15;
#pragma unroll
    for (int r = 0; r < 8; ++r) {
      const int row = wave * 16 + kh * 8 + r;
      Qb[(size_t)(n0 + row) * DDIM + e] = acc[r];
    }
  }
}

// ---------------------------------------------------------------------------
extern "C" void kernel_launch(void* const* d_in, const int* in_sizes, int n_in,
                              void* d_out, int out_size, void* d_ws, size_t ws_size,
                              hipStream_t stream)
{
  const float* query = (const float*)d_in[0];
  const float* key   = (const float*)d_in[1];
  const float* value = (const float*)d_in[2];
  const float* Wq    = (const float*)d_in[3];
  const float* Wk    = (const float*)d_in[4];
  const float* Wv    = (const float*)d_in[5];
  const float* ksc   = (const float*)d_in[6];
  const float* kbi   = (const float*)d_in[7];
  const float* vsc   = (const float*)d_in[8];
  const float* vbi   = (const float*)d_in[9];

  const size_t NTOK = (size_t)BATCH * S_LEN * FDIM;  // 33,554,432 floats
  float* att = (float*)d_out;   // att_output region (also holds Yq, updated in place)
  float* p   = att + NTOK;      // p_attn region

  float* Yk = (float*)d_ws;
  float* Yv = Yk + NTOK;
  float* partials = Yv + NTOK;  // 64 * 8 * 4096 floats

  dim3 gA(FDIM / 128, (BATCH * S_LEN) / 128);  // (8, 256)
  proj_gemm_kernel<<<gA, 256, 0, stream>>>(query, Wq, att);
  proj_gemm_kernel<<<gA, 256, 0, stream>>>(key,   Wk, Yk);
  proj_gemm_kernel<<<gA, 256, 0, stream>>>(value, Wv, Yv);

  ktv_partial_kernel<<<dim3(BATCH * HEADS, NCHUNK), 512, 0, stream>>>(
      Yk, Yv, ksc, kbi, vsc, vbi, partials);

  reduce_pattn_kernel<<<(BATCH * HEADS * DDIM * DDIM) / 256, 256, 0, stream>>>(
      partials, p);

  q_pattn_kernel<<<dim3(S_LEN / 128, BATCH * HEADS), 256, 0, stream>>>(att, p);
}
